// GNN_mol_atom_aggregate_13322988552788
// MI455X (gfx1250) — compile-verified, matrix operand order checked by reading, not run on hardware
//
#include <hip/hip_runtime.h>
#include <hip/hip_bf16.h>

// ---- CDNA5 WMMA vector types (probe-confirmed signatures) ----
typedef __attribute__((ext_vector_type(16))) __bf16 v16bf;
typedef __attribute__((ext_vector_type(8)))  float  v8f;

#define B_N 2048
#define L_N 128
#define D_N 256
#define NEG_SLOPE 0.2f
#define LN_EPS 1e-5f

// LDS layout (floats):
//   nb   [128*256]  f32   (128 KB)
//   A_s  [128*256]  bf16  ( 64 KB -> 16384 floats)
//   q_s  [256], wa_s [256], sc_s [128], red [512]
#define NB_FLOATS   (L_N * D_N)
#define AS_FLOATS   (L_N * D_N / 2)
#define SMEM_FLOATS (NB_FLOATS + AS_FLOATS + D_N + D_N + L_N + 512)
#define SMEM_BYTES  (SMEM_FLOATS * sizeof(float))

// ---------------------------------------------------------------------------
// Kernel 0: convert W1 / W2 (f32, row-major DxD) to bf16 in workspace.
// ---------------------------------------------------------------------------
__global__ void convert_weights(const float* __restrict__ W1,
                                const float* __restrict__ W2,
                                __bf16* __restrict__ W1b,
                                __bf16* __restrict__ W2b) {
    int i = blockIdx.x * blockDim.x + threadIdx.x;
    if (i < D_N * D_N) {
        W1b[i] = (__bf16)W1[i];
        W2b[i] = (__bf16)W2[i];
    }
}

// 16-bit A-fragment (16x32 K-step) from a row-major f32 row (global).
// CDNA5 ISA layout: lanes 0-15 hold K {kb..kb+7, kb+16..kb+23}, kb=ks*32;
// lanes 16-31 the same pattern shifted by +8.
__device__ __forceinline__ v16bf load_a_frag_f32(const float* __restrict__ rowPtr,
                                                 int ks, int laneHi) {
    const int kb = ks * 32 + laneHi * 8;
    const float4 a0 = *(const float4*)(rowPtr + kb);
    const float4 a1 = *(const float4*)(rowPtr + kb + 4);
    const float4 a2 = *(const float4*)(rowPtr + kb + 16);
    const float4 a3 = *(const float4*)(rowPtr + kb + 20);
    v16bf af;
    af[0]=(__bf16)a0.x; af[1]=(__bf16)a0.y; af[2]=(__bf16)a0.z; af[3]=(__bf16)a0.w;
    af[4]=(__bf16)a1.x; af[5]=(__bf16)a1.y; af[6]=(__bf16)a1.z; af[7]=(__bf16)a1.w;
    af[8]=(__bf16)a2.x; af[9]=(__bf16)a2.y; af[10]=(__bf16)a2.z; af[11]=(__bf16)a2.w;
    af[12]=(__bf16)a3.x; af[13]=(__bf16)a3.y; af[14]=(__bf16)a3.z; af[15]=(__bf16)a3.w;
    return af;
}

// 16-bit fragment from a row-major bf16 row (works for LDS A rows and for
// B = W^T where B[k][n] = W[n][k], lane owning column n). Two 16-byte loads.
__device__ __forceinline__ v16bf load_frag_bf16(const __bf16* __restrict__ row,
                                                int ks, int laneHi) {
    const int kb = ks * 32 + laneHi * 8;
    union { v16bf v; uint4 u[2]; } fu;
    fu.u[0] = *(const uint4*)(row + kb);
    fu.u[1] = *(const uint4*)(row + kb + 16);
    return fu.v;
}

// ---------------------------------------------------------------------------
// Kernel 1: q = mol_atom_feature @ W1^T + b1   ([2048,256], WMMA bf16)
// One block per 16-row M-tile; 8 waves each own 2 N-tiles.
// ---------------------------------------------------------------------------
__global__ __launch_bounds__(256)
void q_gemm(const float* __restrict__ mol,
            const float* __restrict__ b1,
            const __bf16* __restrict__ W1b,
            float* __restrict__ qbuf) {
    const int tid  = threadIdx.x;
    const int wave = tid >> 5;
    const int lane = tid & 31;
    const int ln16 = lane & 15;
    const int hi   = lane >> 4;
    const int mt   = blockIdx.x;               // 0..127
    const int rowA = mt * 16 + ln16;
    const float* Abase = mol + (size_t)rowA * D_N;

    v16bf afrag[8];
#pragma unroll
    for (int ks = 0; ks < 8; ++ks) afrag[ks] = load_a_frag_f32(Abase, ks, hi);

#pragma unroll
    for (int t = 0; t < 2; ++t) {
        const int nt   = wave * 2 + t;
        const int ncol = nt * 16 + ln16;
        const __bf16* Bbase = W1b + (size_t)ncol * D_N;
        v8f acc = {};
#pragma unroll
        for (int ks = 0; ks < 8; ++ks) {
            v16bf bf = load_frag_bf16(Bbase, ks, hi);
            acc = __builtin_amdgcn_wmma_f32_16x16x32_bf16(
                false, afrag[ks], false, bf, (short)0, acc, false, false);
        }
        const float bias = b1[ncol];
#pragma unroll
        for (int r = 0; r < 8; ++r) {
            const int m = r + hi * 8;                    // C layout: M = vgpr + 8*laneHi
            qbuf[(size_t)(mt * 16 + m) * D_N + ncol] = acc[r] + bias;
        }
    }
}

// ---------------------------------------------------------------------------
// Kernel 2: fused per-molecule GEMM + GATv2 score + softmax + context + LN.
// One block (8 waves) per molecule b.
//   - atom[b] staged to LDS as bf16 (read once from HBM)
//   - each wave owns 2 N-tiles for ALL 8 M-tiles; its 16 B-fragments of W2
//     live in registers, so W2 is fetched exactly once per workgroup.
// ---------------------------------------------------------------------------
__global__ __launch_bounds__(256)
void fused_kernel(const float* __restrict__ atom,     // [B,L,D]
                  const float* __restrict__ attend,   // [B,L]
                  const float* __restrict__ smask,    // [B,L]
                  const float* __restrict__ b2,       // [D]
                  const float* __restrict__ w_align,  // [D]
                  const float* __restrict__ b_align,  // [1]
                  const float* __restrict__ gamma_,   // [D]
                  const float* __restrict__ beta_,    // [D]
                  const float* __restrict__ qbuf,     // [B,D] (ws)
                  const __bf16* __restrict__ W2b,     // [D,D] bf16 (ws)
                  float* __restrict__ out) {          // [B,D]
    extern __shared__ float smem[];
    float*  nb   = smem;                       // 128*256 f32
    __bf16* A_s  = (__bf16*)(nb + NB_FLOATS);  // 128*256 bf16
    float*  q_s  = nb + NB_FLOATS + AS_FLOATS; // 256
    float*  wa_s = q_s + D_N;                  // 256
    float*  sc_s = wa_s + D_N;                 // 128
    float*  red  = sc_s + L_N;                 // 512

    const int b    = blockIdx.x;
    const int tid  = threadIdx.x;
    const int wave = tid >> 5;
    const int lane = tid & 31;
    const int ln16 = lane & 15;
    const int hi   = lane >> 4;

    // stage q[b] and w_align
    q_s[tid]  = qbuf[(size_t)b * D_N + tid];
    wa_s[tid] = w_align[tid];

    // ---- stage atom[b] -> LDS bf16 (cooperative, coalesced 32B chunks) ----
    const float* Ag = atom + (size_t)b * L_N * D_N;
    for (int i = tid * 8; i < L_N * D_N; i += 256 * 8) {
        const float4 f0 = *(const float4*)(Ag + i);
        const float4 f1 = *(const float4*)(Ag + i + 4);
        union { uint4 u; __bf16 h[8]; } pk;
        pk.h[0]=(__bf16)f0.x; pk.h[1]=(__bf16)f0.y; pk.h[2]=(__bf16)f0.z; pk.h[3]=(__bf16)f0.w;
        pk.h[4]=(__bf16)f1.x; pk.h[5]=(__bf16)f1.y; pk.h[6]=(__bf16)f1.z; pk.h[7]=(__bf16)f1.w;
        *(uint4*)(A_s + i) = pk.u;
    }

    // ---- load this wave's 16 B-fragments of W2^T (held in VGPRs) ----
    v16bf bfrag[2][8];
    float bias[2];
#pragma unroll
    for (int t = 0; t < 2; ++t) {
        const int ncol = (wave * 2 + t) * 16 + ln16;
        const __bf16* Bbase = W2b + (size_t)ncol * D_N;
#pragma unroll
        for (int ks = 0; ks < 8; ++ks)
            bfrag[t][ks] = load_frag_bf16(Bbase, ks, hi);
        bias[t] = b2[ncol];
    }
    __syncthreads();

    // ---- GEMM: nb = atom[b] @ W2^T + b2 (A from LDS, B from registers) ----
#pragma unroll 1
    for (int mt = 0; mt < 8; ++mt) {
        const __bf16* Arow = A_s + (size_t)(mt * 16 + ln16) * D_N;
        v16bf afrag[8];
#pragma unroll
        for (int ks = 0; ks < 8; ++ks) afrag[ks] = load_frag_bf16(Arow, ks, hi);

#pragma unroll
        for (int t = 0; t < 2; ++t) {
            const int ncol = (wave * 2 + t) * 16 + ln16;
            v8f acc = {};
#pragma unroll
            for (int ks = 0; ks < 8; ++ks) {
                acc = __builtin_amdgcn_wmma_f32_16x16x32_bf16(
                    false, afrag[ks], false, bfrag[t][ks], (short)0, acc, false, false);
            }
#pragma unroll
            for (int r = 0; r < 8; ++r) {
                const int row = mt * 16 + r + hi * 8;   // C layout: M = vgpr + 8*laneHi
                nb[row * D_N + ncol] = acc[r] + bias[t];
            }
        }
    }
    __syncthreads();

    // ---- score[l] = leaky_relu(q + nb[l]) . w_align + b_align + mask ----
    const float balign = b_align[0];
    for (int i = 0; i < 16; ++i) {
        const int l = wave * 16 + i;
        const float* rp = nb + l * D_N + lane * 8;
        float p = 0.f;
#pragma unroll
        for (int j = 0; j < 8; ++j) {
            const int d = lane * 8 + j;
            const float v = q_s[d] + rp[j];
            const float a = (v >= 0.f) ? v : NEG_SLOPE * v;
            p += a * wa_s[d];
        }
#pragma unroll
        for (int off = 16; off > 0; off >>= 1) p += __shfl_xor(p, off, 32);
        if (lane == 0)
            sc_s[l] = p + balign + smask[(size_t)b * L_N + l];
    }
    __syncthreads();

    // ---- softmax over L, times attend mask ----
    float mx = -3.0e38f;
    for (int l = 0; l < L_N; ++l) mx = fmaxf(mx, sc_s[l]);
    float e = 0.f;
    if (tid < L_N) e = __expf(sc_s[tid] - mx);
    __syncthreads();
    if (tid < L_N) sc_s[tid] = e;
    __syncthreads();
    float den = 0.f;
    for (int l = 0; l < L_N; ++l) den += sc_s[l];
    const float inv = 1.0f / den;
    __syncthreads();
    if (tid < L_N)
        sc_s[tid] = sc_s[tid] * inv * attend[(size_t)b * L_N + tid];
    __syncthreads();

    // ---- ctx[d] = sum_l attn[l] * nb[l][d]  (thread d = tid) ----
    float c = 0.f;
    for (int l = 0; l < L_N; ++l) c += sc_s[l] * nb[l * D_N + tid];

    // ---- LayerNorm over D via block reduction ----
    red[tid]       = c;
    red[D_N + tid] = c * c;
    __syncthreads();
    for (int s = 128; s > 0; s >>= 1) {
        if (tid < s) {
            red[tid]       += red[tid + s];
            red[D_N + tid] += red[D_N + tid + s];
        }
        __syncthreads();
    }
    const float mu  = red[0] * (1.0f / D_N);
    const float var = red[D_N] * (1.0f / D_N) - mu * mu;
    const float rs  = rsqrtf(var + LN_EPS);
    out[(size_t)b * D_N + tid] = (c - mu) * rs * gamma_[tid] + beta_[tid];
}

// ---------------------------------------------------------------------------
extern "C" void kernel_launch(void* const* d_in, const int* in_sizes, int n_in,
                              void* d_out, int out_size, void* d_ws, size_t ws_size,
                              hipStream_t stream) {
    const float* mol     = (const float*)d_in[0];   // [B,D]
    const float* atom    = (const float*)d_in[1];   // [B,L,D]
    const float* attend  = (const float*)d_in[2];   // [B,L,1]
    const float* smask   = (const float*)d_in[3];   // [B,L,1]
    const float* W1      = (const float*)d_in[4];   // [D,D]
    const float* b1      = (const float*)d_in[5];   // [D]
    const float* W2      = (const float*)d_in[6];   // [D,D]
    const float* b2      = (const float*)d_in[7];   // [D]
    const float* w_align = (const float*)d_in[8];   // [D,1]
    const float* b_align = (const float*)d_in[9];   // [1]
    const float* gamma_  = (const float*)d_in[10];  // [D]
    const float* beta_   = (const float*)d_in[11];  // [D]
    float* out = (float*)d_out;

    char* ws = (char*)d_ws;
    float*  qbuf = (float*)ws;                                    // 2 MB
    __bf16* W2b  = (__bf16*)(ws + (size_t)2 * 1024 * 1024);       // 128 KB
    __bf16* W1b  = (__bf16*)(ws + (size_t)2 * 1024 * 1024 + 128 * 1024);

    convert_weights<<<(D_N * D_N + 255) / 256, 256, 0, stream>>>(W1, W2, W1b, W2b);
    q_gemm<<<B_N / 16, 256, 0, stream>>>(mol, b1, W1b, qbuf);
    fused_kernel<<<B_N, 256, SMEM_BYTES, stream>>>(
        atom, attend, smask, b2, w_align, b_align, gamma_, beta_, qbuf, W2b, out);
}
